// SequenceEncoder_69063074120430
// MI455X (gfx1250) — matrix operand order repaired and checked by reading, not compiled, not run on hardware
//
#include <hip/hip_runtime.h>

// ---------------------------------------------------------------------------
// SequenceEncoder on MI455X (gfx1250), reassociated around HID=100:
//   G = h^T h    (bf16 WMMA, f32 acc)   [per batch, pad 100->128]
//   A2T = (Wq^T G)^T, c = bq G          (scalar f32, ~80 MFLOP total)
//   out = h A2 + c                      (bf16 WMMA, f32 acc)
// Naive reference = 54 GFLOP + 536 MB scores tensor; this = 5.4 GFLOP,
// ~100 MB traffic -> memory-bound, so bf16 WMMA keeps compute off the
// critical path while f32 accumulation bounds error at ~0.3%.
// ---------------------------------------------------------------------------

typedef __attribute__((ext_vector_type(16))) __bf16 v16bf;
typedef __attribute__((ext_vector_type(8)))  float  v8f;

#define BATCH 32
#define SEQ   2048
#define HID   100
#define DP    128   // hidden dim padded to multiple of 32 (zeros beyond 100)

// ---- WMMA fragment loaders (contiguous k order shared by A and B, wave32) ---------

// A-matrix 16x32 bf16, row-major source with row stride `stride`.
// lane<16: row=lane,   elems 0..15 = K {0..15}
// lane>=16: row=lane-16, elems 0..15 = K {16..31}
__device__ __forceinline__ v16bf load_a_frag(const __bf16* __restrict__ base,
                                             int stride, int row0, int k0, int lane) {
    int r    = row0 + (lane & 15);
    int koff = (lane >> 4) * 16;
    const __bf16* p = base + (long)r * stride + k0 + koff;
    v16bf a;
#pragma unroll
    for (int t = 0; t < 16; ++t) a[t] = p[t];
    return a;
}

// B-matrix 32x16 bf16 where B[k][n] = T[col0+n][k] (T row-major, stride `stride`).
// lane<16: col=lane, elems 0..15 = K {k0..k0+15}; lane>=16: K {k0+16..k0+31}
__device__ __forceinline__ v16bf load_b_fragT(const __bf16* __restrict__ baseT,
                                              int stride, int col0, int k0, int lane) {
    int n  = col0 + (lane & 15);
    int kb = k0 + (lane >> 4) * 16;
    const __bf16* p = baseT + (long)n * stride + kb;
    v16bf b;
#pragma unroll
    for (int t = 0; t < 16; ++t) b[t] = p[t];
    return b;
}

__device__ __forceinline__ v8f wmma_bf16(v16bf a, v16bf b, v8f c) {
    return __builtin_amdgcn_wmma_f32_16x16x32_bf16(false, a, false, b,
                                                   (short)0, c, false, false);
}

// ---- K1: embedding gather -> h (bf16, row-major, padded) and h^T ----------
// One block = 16 sequence rows of one batch. 128 threads.
__global__ void k_gather(const int* __restrict__ x, const float* __restrict__ embed,
                         __bf16* __restrict__ h, __bf16* __restrict__ hT) {
    __shared__ __bf16 tile[16][132];           // padded row dodges LDS bank conflicts
    const int bid = blockIdx.x;
    const int b   = bid >> 7;                  // SEQ/16 = 128 blocks per batch
    const int n0  = (bid & 127) << 4;
    const int tid = threadIdx.x;

#pragma unroll 4
    for (int i = tid; i < 16 * DP; i += 128) {
        const int r = i >> 7;                  // uniform per iteration -> scalar idx load
        const int d = i & 127;
        const int idx = x[b * SEQ + n0 + r];
        const float v = (d < HID) ? embed[(long)idx * HID + d] : 0.0f;
        const __bf16 bv = (__bf16)v;
        tile[r][d] = bv;
        h[((long)b * SEQ + n0 + r) * DP + d] = bv;       // coalesced
    }
    __syncthreads();
#pragma unroll 4
    for (int i = tid; i < DP * 16; i += 128) {
        const int d  = i >> 4;
        const int nn = i & 15;
        hT[((long)b * DP + d) * SEQ + n0 + nn] = tile[nn][d];  // 32B runs per 16 lanes
    }
}

// ---- K2: G[b] = h^T h via v_wmma_f32_16x16x32_bf16 ------------------------
// Register-blocked 16x32 strip per wave: A fragment reused across 2 B tiles.
// 32 batches x 8 o-tiles x 4 m-pairs = 1024 waves.
__global__ void k_gram(const __bf16* __restrict__ hT, float* __restrict__ G) {
    const int w    = blockIdx.x * (blockDim.x >> 5) + (threadIdx.x >> 5);
    const int lane = threadIdx.x & 31;
    const int b  = w >> 5;
    const int t  = w & 31;
    const int o0 = (t >> 2) << 4;      // 8 row tiles
    const int m0 = (t & 3) << 5;       // 4 column pairs (32 cols each)
    const __bf16* base = hT + (long)b * DP * SEQ;

    v8f acc0 = {}, acc1 = {};
    for (int k0 = 0; k0 < SEQ; k0 += 32) {
        v16bf a  = load_a_frag (base, SEQ, o0, k0, lane);      // rows of h^T
        v16bf b0 = load_b_fragT(base, SEQ, m0,      k0, lane); // cols of h = rows of h^T
        v16bf b1 = load_b_fragT(base, SEQ, m0 + 16, k0, lane);
        acc0 = wmma_bf16(a, b0, acc0);
        acc1 = wmma_bf16(a, b1, acc1);
    }
    float* g = G + (long)b * DP * DP;
    const int rbase = (lane >> 4) << 3;
    const int col   = lane & 15;
#pragma unroll
    for (int r = 0; r < 8; ++r) {
        g[(o0 + r + rbase) * DP + m0      + col] = acc0[r];
        g[(o0 + r + rbase) * DP + m0 + 16 + col] = acc1[r];
    }
}

// ---- K3: fold Wq/bq through G (tiny, scalar f32) --------------------------
// A2T[m][d] = sum_o Wq[o][d] * G[o][m]  (stored transposed for K4's B-frags)
// c[m]      = sum_o bq[o]    * G[o][m]
__global__ void k_fold(const float* __restrict__ Wq, const float* __restrict__ bq,
                       const float* __restrict__ G, __bf16* __restrict__ A2T,
                       float* __restrict__ c) {
    const int b = blockIdx.x, tid = threadIdx.x;   // 256 threads
    const float* g = G + (long)b * DP * DP;

    for (int i = tid; i < DP * DP; i += 256) {
        const int m = i >> 7;
        const int d = i & 127;
        float acc = 0.0f;
        if (d < HID) {
#pragma unroll 4
            for (int o = 0; o < HID; ++o) acc += Wq[o * HID + d] * g[o * DP + m];
        }
        A2T[(long)b * DP * DP + m * DP + d] = (__bf16)acc;
    }
    for (int m = tid; m < DP; m += 256) {
        float acc = 0.0f;
        for (int o = 0; o < HID; ++o) acc += bq[o] * g[o * DP + m];
        c[b * DP + m] = acc;
    }
}

// ---- K4: out[b] = h[b] @ A2[b] + c[b] via WMMA ----------------------------
// Register-blocked 32x16 per wave: B fragment (from tiny A2T) reused across
// 2 A tiles. 32 batches x 64 row-pairs x 7 col tiles = 14336 waves.
__global__ void k_out(const __bf16* __restrict__ h, const __bf16* __restrict__ A2T,
                      const float* __restrict__ c, float* __restrict__ out) {
    const int w    = blockIdx.x * (blockDim.x >> 5) + (threadIdx.x >> 5);
    const int lane = threadIdx.x & 31;
    const int ct  = w % 7;
    const int tmp = w / 7;
    const int rp  = tmp & 63;
    const int b   = tmp >> 6;
    const int n0 = rp << 5;            // 32 rows per wave
    const int m0 = ct << 4;
    const __bf16* hb = h   + (long)b * SEQ * DP;
    const __bf16* at = A2T + (long)b * DP * DP;

    v8f acc0 = {}, acc1 = {};
#pragma unroll
    for (int k0 = 0; k0 < DP; k0 += 32) {
        v16bf bb = load_b_fragT(at, DP, m0, k0, lane);
        v16bf a0 = load_a_frag (hb, DP, n0,      k0, lane);
        v16bf a1 = load_a_frag (hb, DP, n0 + 16, k0, lane);
        acc0 = wmma_bf16(a0, bb, acc0);
        acc1 = wmma_bf16(a1, bb, acc1);
    }
    const int col = m0 + (lane & 15);
    if (col < HID) {
        const float bias = c[b * DP + col];
#pragma unroll
        for (int r = 0; r < 8; ++r) {
            const int row = n0 + r + ((lane >> 4) << 3);
            out[((long)b * SEQ + row)      * HID + col] = acc0[r] + bias;
            out[((long)b * SEQ + row + 16) * HID + col] = acc1[r] + bias;
        }
    }
}

// ---------------------------------------------------------------------------
extern "C" void kernel_launch(void* const* d_in, const int* in_sizes, int n_in,
                              void* d_out, int out_size, void* d_ws, size_t ws_size,
                              hipStream_t stream) {
    const int*   x     = (const int*)  d_in[0];
    const float* embed = (const float*)d_in[1];
    const float* Wq    = (const float*)d_in[2];
    const float* bq    = (const float*)d_in[3];
    float*       out   = (float*)d_out;

    char* ws = (char*)d_ws;
    size_t off = 0;
    __bf16* hT  = (__bf16*)(ws + off); off += (size_t)BATCH * DP  * SEQ * sizeof(__bf16);
    __bf16* h   = (__bf16*)(ws + off); off += (size_t)BATCH * SEQ * DP  * sizeof(__bf16);
    float*  G   = (float*) (ws + off); off += (size_t)BATCH * DP  * DP  * sizeof(float);
    __bf16* A2T = (__bf16*)(ws + off); off += (size_t)BATCH * DP  * DP  * sizeof(__bf16);
    float*  cv  = (float*) (ws + off);

    // K1: gather + bf16 convert + transpose (both layouts coalesced via LDS)
    k_gather<<<BATCH * (SEQ / 16), 128, 0, stream>>>(x, embed, h, hT);
    // K2: 1024 waves, 8 waves/block
    k_gram<<<(BATCH * 32) / 8, 256, 0, stream>>>(hT, G);
    // K3: tiny per-batch fold
    k_fold<<<BATCH, 256, 0, stream>>>(Wq, bq, G, A2T, cv);
    // K4: 14336 waves, 8 waves/block
    k_out<<<(BATCH * 64 * 7) / 8, 256, 0, stream>>>(h, A2T, cv, out);
}
